// _TinyMoELayer_66279935312319
// MI455X (gfx1250) — matrix operand order repaired and checked by reading, not compiled, not run on hardware
//
#include <hip/hip_runtime.h>
#include <hip/hip_bf16.h>

// ---------------------------------------------------------------------------
// TinyMoE for MI455X (gfx1250): dense-expert-compute / sparse-combine MoE.
//   1) cast hidden states f32->bf16                       (ws: Xb)
//   2) transpose+cast w1,w2 to [E][N][K] bf16 (K-contig)  (ws: W1t, W2t)
//   3) router: logits -> top2 -> combine[T][8] f32        (ws: Combine)
//   4) per expert e: GEMM1 relu(X@W1+b1) -> Hdn (bf16 ws), then
//      GEMM2 (Hdn@W2+b2) * combine[:,e] accumulated into d_out (f32).
// GEMMs: 128x128x32 tiles, 8 wave32/WG, 2x4 WMMA 16x16 tiles per wave with
// v_wmma_f32_16x16x32_bf16, LDS double buffering, async global->LDS copies
// (ASYNCcnt) via global_load_async_to_lds_b128.
// ---------------------------------------------------------------------------

#define T_TOK 16384
#define HDIM  1024
#define HDIM2 2048
#define NEXP  8

typedef __bf16        v16bf __attribute__((ext_vector_type(16)));
typedef float         v8f   __attribute__((ext_vector_type(8)));
typedef unsigned int  u32x4 __attribute__((ext_vector_type(4)));

// Async-copy builtin takes typed 16B-vector pointers in AS(1)/AS(3).
typedef int moe_v4i __attribute__((vector_size(16)));
typedef __attribute__((address_space(1))) moe_v4i moe_gv4i;  // global
typedef __attribute__((address_space(3))) moe_v4i moe_sv4i;  // LDS

#if __has_builtin(__builtin_amdgcn_global_load_async_to_lds_b128)
#define MOE_ASYNC_COPY 1
#else
#define MOE_ASYNC_COPY 0
#endif

__device__ __forceinline__ void copy16_g2s(unsigned short* dst, const __bf16* src) {
#if MOE_ASYNC_COPY
  __builtin_amdgcn_global_load_async_to_lds_b128(
      (moe_gv4i*)src, (moe_sv4i*)dst, 0, 0);
#else
  *(u32x4*)dst = *(const u32x4*)src;
#endif
}

__device__ __forceinline__ void wait_g2s() {
#if MOE_ASYNC_COPY && __has_builtin(__builtin_amdgcn_s_wait_asynccnt)
  __builtin_amdgcn_s_wait_asynccnt(0);
#endif
}

// ------------------------------ converts -----------------------------------

__global__ __launch_bounds__(256) void moe_cvt_bf16(const float* __restrict__ in,
                                                    __bf16* __restrict__ out,
                                                    long n) {
  long i = (long)blockIdx.x * blockDim.x + threadIdx.x;
  if (i < n) out[i] = (__bf16)in[i];
}

// in: [E][R][C] f32 row-major  ->  out: [E][C][R] bf16 (K-contiguous for GEMM B)
__global__ __launch_bounds__(256) void moe_transpose_cvt(const float* __restrict__ in,
                                                         __bf16* __restrict__ out,
                                                         int R, int C, long total) {
  long i = (long)blockIdx.x * blockDim.x + threadIdx.x;
  if (i >= total) return;
  long RC  = (long)R * C;
  int  e   = (int)(i / RC);
  long rem = i - (long)e * RC;
  int  n   = (int)(rem / R);
  int  k   = (int)(rem - (long)n * R);
  out[i] = (__bf16)in[(long)e * RC + (long)k * C + n];
}

// ------------------------------- router ------------------------------------

__global__ __launch_bounds__(256) void moe_router(const float* __restrict__ X,
                                                  const float* __restrict__ GW,
                                                  float* __restrict__ C) {
  const int lane = threadIdx.x & 31;
  const int wv   = threadIdx.x >> 5;
  const int t    = blockIdx.x * 8 + wv;
  if (t >= T_TOK) return;

  float acc[NEXP];
#pragma unroll
  for (int e = 0; e < NEXP; ++e) acc[e] = 0.f;

  for (int h = lane; h < HDIM; h += 32) {
    float x = X[(size_t)t * HDIM + h];
    const float* g = GW + (size_t)h * NEXP;
#pragma unroll
    for (int e = 0; e < NEXP; ++e) acc[e] += x * g[e];
  }
#pragma unroll
  for (int off = 16; off > 0; off >>= 1)
#pragma unroll
    for (int e = 0; e < NEXP; ++e) acc[e] += __shfl_xor(acc[e], off, 32);

  if (lane == 0) {
    int i1 = 0;
#pragma unroll
    for (int e = 1; e < NEXP; ++e) if (acc[e] > acc[i1]) i1 = e;
    int i2 = -1; float m2 = -3.4e38f;
#pragma unroll
    for (int e = 0; e < NEXP; ++e) if (e != i1 && acc[e] > m2) { m2 = acc[e]; i2 = e; }
    // top-2 softmax renorm: w1 = 1/(1+exp(l2-l1)), w2 = 1-w1
    float w2 = __expf(m2 - acc[i1]);
    float s  = 1.f + w2;
#pragma unroll
    for (int e = 0; e < NEXP; ++e) C[(size_t)t * NEXP + e] = 0.f;
    C[(size_t)t * NEXP + i1] = 1.f / s;
    C[(size_t)t * NEXP + i2] = w2 / s;
  }
}

// ------------------------------- GEMM --------------------------------------

#define BM 128
#define BN 128
#define BK 32
#define LDSROW 40                    // BK + 8 bf16 pad -> 80B rows (16B aligned)
#define BUF_ELEMS (2 * BM * LDSROW)  // A tile + B tile, bf16 elements

union FragAB { u32x4 q[2]; v16bf v; };

template <int MODE> // 0: GEMM1 relu->bf16 | 1: GEMM2 overwrite | 2: GEMM2 accumulate
__global__ __launch_bounds__(256)
void moe_gemm(const __bf16* __restrict__ A,   // [M x K] row-major
              const __bf16* __restrict__ Bt,  // [N x K] row-major (pre-transposed)
              const float*  __restrict__ bias,// [N]
              __bf16*       __restrict__ outb,// MODE 0: [M x N] bf16
              float*        __restrict__ outf,// MODE 1/2: [M x N] f32
              const float*  __restrict__ comb,// [M x NEXP]
              int expert, int K, int N)
{
  __shared__ unsigned short smem[2 * BUF_ELEMS];

  const int tid  = threadIdx.x;
  const int lane = tid & 31;
  const int wv   = tid >> 5;
  const int wm   = (wv & 3) * 32;   // wave M offset inside 128-tile
  const int wn   = (wv >> 2) * 64;  // wave N offset inside 128-tile
  const int lg   = lane >> 4;       // K-half select per WMMA 16-bit layout
  const int lr   = lane & 15;

  const int m0 = blockIdx.y * BM;
  const int n0 = blockIdx.x * BN;
  const int nk = K / BK;

  auto stage = [&](int buf, int kk) {
    unsigned short* as = smem + buf * BUF_ELEMS;
    unsigned short* bs = as + BM * LDSROW;
    const __bf16* ag = A  + (size_t)m0 * K + (size_t)kk * BK;
    const __bf16* bg = Bt + (size_t)n0 * K + (size_t)kk * BK;
#pragma unroll
    for (int t = 0; t < 2; ++t) {
      int c = tid + t * 256;        // 512 x 16B chunks per 128x32 bf16 tile
      int r = c >> 2;
      int q = c & 3;
      copy16_g2s(as + r * LDSROW + q * 8, ag + (size_t)r * K + q * 8);
      copy16_g2s(bs + r * LDSROW + q * 8, bg + (size_t)r * K + q * 8);
    }
  };

  v8f acc[2][4];
#pragma unroll
  for (int i = 0; i < 2; ++i)
#pragma unroll
    for (int j = 0; j < 4; ++j)
      acc[i][j] = (v8f){0.f, 0.f, 0.f, 0.f, 0.f, 0.f, 0.f, 0.f};

  stage(0, 0);

  for (int kk = 0; kk < nk; ++kk) {
    wait_g2s();
    __syncthreads();
    if (kk + 1 < nk) stage((kk + 1) & 1, kk + 1); // writes the *other* buffer

    const unsigned short* as = smem + (kk & 1) * BUF_ELEMS;
    const unsigned short* bs = as + BM * LDSROW;

    FragAB a[2], b[4];
#pragma unroll
    for (int i = 0; i < 2; ++i) {
      // A 16x32 bf16: lanes 0-15 hold K {0..7,16..23}, lanes 16-31 {8..15,24..31}
      const unsigned short* p = as + (wm + i * 16 + lr) * LDSROW + lg * 8;
      a[i].q[0] = *(const u32x4*)(p);
      a[i].q[1] = *(const u32x4*)(p + 16);
    }
#pragma unroll
    for (int j = 0; j < 4; ++j) {
      // B 32x16 bf16 (stored [N][K]): lanes 0-15 K {0..7,16..23}, lanes 16-31 {8..15,24..31}
      const unsigned short* p = bs + (wn + j * 16 + lr) * LDSROW + lg * 8;
      b[j].q[0] = *(const u32x4*)(p);
      b[j].q[1] = *(const u32x4*)(p + 16);
    }
#pragma unroll
    for (int i = 0; i < 2; ++i)
#pragma unroll
      for (int j = 0; j < 4; ++j)
        acc[i][j] = __builtin_amdgcn_wmma_f32_16x16x32_bf16(
            false, a[i].v, false, b[j].v, (short)0, acc[i][j], false, false);
  }

  // Epilogue. D layout: VGPR r -> M = r + 8*(lane>=16); N = lane&15.
#pragma unroll
  for (int i = 0; i < 2; ++i) {
#pragma unroll
    for (int j = 0; j < 4; ++j) {
      const int nn = n0 + wn + j * 16 + lr;
      const int mb = m0 + wm + i * 16 + lg * 8;
      const float bv = bias[nn];
#pragma unroll
      for (int r = 0; r < 8; ++r) {
        const int m = mb + r;
        float v = acc[i][j][r] + bv;
        if (MODE == 0) {
          v = v > 0.f ? v : 0.f;
          outb[(size_t)m * N + nn] = (__bf16)v;
        } else {
          const float w = comb[(size_t)m * NEXP + expert];
          const size_t o = (size_t)m * N + nn;
          if (MODE == 1) outf[o] = w * v;
          else           outf[o] += w * v;
        }
      }
    }
  }
}

// ------------------------------ launcher -----------------------------------

extern "C" void kernel_launch(void* const* d_in, const int* in_sizes, int n_in,
                              void* d_out, int out_size, void* d_ws, size_t ws_size,
                              hipStream_t stream) {
  (void)in_sizes; (void)n_in; (void)out_size; (void)ws_size;

  const float* X  = (const float*)d_in[0]; // [T, H]
  const float* GW = (const float*)d_in[1]; // [H, E]
  const float* W1 = (const float*)d_in[2]; // [E, H, 2H]
  const float* B1 = (const float*)d_in[3]; // [E, 2H]
  const float* W2 = (const float*)d_in[4]; // [E, 2H, H]
  const float* B2 = (const float*)d_in[5]; // [E, H]
  float* Out = (float*)d_out;              // [T, H]

  char* w = (char*)d_ws;
  size_t off = 0;
  __bf16* Xb  = (__bf16*)(w + off); off += (size_t)T_TOK * HDIM  * 2;
  __bf16* W1t = (__bf16*)(w + off); off += (size_t)NEXP * HDIM2 * HDIM * 2;
  __bf16* W2t = (__bf16*)(w + off); off += (size_t)NEXP * HDIM  * HDIM2 * 2;
  __bf16* Hdn = (__bf16*)(w + off); off += (size_t)T_TOK * HDIM2 * 2;
  float* Combine = (float*)(w + off);

  // 1) casts / transposes
  {
    long n = (long)T_TOK * HDIM;
    moe_cvt_bf16<<<dim3((unsigned)((n + 255) / 256)), dim3(256), 0, stream>>>(X, Xb, n);
  }
  {
    long tot = (long)NEXP * HDIM * HDIM2;
    moe_transpose_cvt<<<dim3((unsigned)((tot + 255) / 256)), dim3(256), 0, stream>>>(
        W1, W1t, HDIM, HDIM2, tot);   // [E][H][2H] -> [E][2H][H]
    moe_transpose_cvt<<<dim3((unsigned)((tot + 255) / 256)), dim3(256), 0, stream>>>(
        W2, W2t, HDIM2, HDIM, tot);   // [E][2H][H] -> [E][H][2H]
  }

  // 2) router
  moe_router<<<dim3(T_TOK / 8), dim3(256), 0, stream>>>(X, GW, Combine);

  // 3) experts: dense compute, sparse combine
  const dim3 blk(256);
  const dim3 g1(HDIM2 / BN, T_TOK / BM);  // GEMM1: M=T, N=2H, K=H
  const dim3 g2(HDIM  / BN, T_TOK / BM);  // GEMM2: M=T, N=H,  K=2H
  for (int e = 0; e < NEXP; ++e) {
    const __bf16* w1e = W1t + (size_t)e * HDIM2 * HDIM;
    const __bf16* w2e = W2t + (size_t)e * HDIM  * HDIM2;
    moe_gemm<0><<<g1, blk, 0, stream>>>(Xb, w1e, B1 + (size_t)e * HDIM2,
                                        Hdn, nullptr, nullptr, e, HDIM, HDIM2);
    if (e == 0)
      moe_gemm<1><<<g2, blk, 0, stream>>>(Hdn, w2e, B2 + (size_t)e * HDIM,
                                          nullptr, Out, Combine, e, HDIM2, HDIM);
    else
      moe_gemm<2><<<g2, blk, 0, stream>>>(Hdn, w2e, B2 + (size_t)e * HDIM,
                                          nullptr, Out, Combine, e, HDIM2, HDIM);
  }
}
